// SwinTransformerBlock_40321152975590
// MI455X (gfx1250) — matrix operand order repaired and checked by reading, not compiled, not run on hardware
//
#include <hip/hip_runtime.h>
#include <hip/hip_bf16.h>

typedef __attribute__((ext_vector_type(16))) _Float16 v16h;
typedef __attribute__((ext_vector_type(8)))  float    v8f;

#define WMMA_F16(A, B, C) \
  __builtin_amdgcn_wmma_f32_16x16x32_f16(false, (A), false, (B), (short)0, (C), false, false)

// Problem constants (fixed by the reference)
constexpr int C_DIM   = 128;
constexpr int NH      = 4;
constexpr int WS      = 7;
constexpr int N_TOK   = 49;   // tokens per window
constexpr int HW      = 112;
constexpr int NWX     = 16;   // windows per row/col
constexpr int NWIN    = 256;  // windows per image
constexpr float QSCALE = 0.17677669529663687f; // 1/sqrt(32)

// LDS layout (bytes)
constexpr int SX_OFF  = 0;        // 64x128 f16 : x window (reused as attn-out A operand)
constexpr int SQ_OFF  = 16384;    // 64x128 f16 : q (scaled)
constexpr int SK_OFF  = 32768;    // 64x128 f16 : k
constexpr int SVT_OFF = 49152;    // 128x64 f16 : v transposed [channel][token]
constexpr int SS_OFF  = 65536;    // 4 x 64x64 f32 : scores per head
constexpr int SP_OFF  = 131072;   // 4 x 64x64 f16 : probs per head
constexpr int SMEM_BYTES = 163840;

// ---- fragment loaders (CDNA5 wave32 WMMA layouts, CDNA5 ISA §7.12.2) ----
// A (16x32 f16): lane m = l&15, s = l>>4; element e -> K = 16*s + e (matches B)
__device__ __forceinline__ v16h load_a_frag(const _Float16* p, int stride, int lane) {
  const int m = lane & 15, s = lane >> 4;
  const _Float16* row = p + m * stride + 16 * s;
  v16h a;
#pragma unroll
  for (int vg = 0; vg < 8; ++vg) {
    const int k = 2 * vg;
    a[2 * vg]     = row[k];
    a[2 * vg + 1] = row[k + 1];
  }
  return a;
}

// B (32x16 f16): lane n = l&15, s = l>>4; element e -> K = 16*s + e (16 contiguous f16)
__device__ __forceinline__ v16h load_b_frag(const _Float16* p, int stride, int lane) {
  const int n = lane & 15, s = lane >> 4;
  return *(const v16h*)(p + n * stride + 16 * s);
}

// ---- tiny prep: fp32 weights -> f16 in workspace ----
__global__ void prep_weights(const float* __restrict__ qkv_w,
                             const float* __restrict__ proj_w,
                             _Float16* __restrict__ w16) {
  const int i = blockIdx.x * blockDim.x + threadIdx.x;
  if (i < 384 * 128) {
    w16[i] = (_Float16)qkv_w[i];
  } else if (i < 384 * 128 + 128 * 128) {
    w16[i] = (_Float16)proj_w[i - 384 * 128];
  }
}

// ---- fused Swin block: one workgroup (128 threads = 4 waves) per window ----
__global__ __launch_bounds__(128)
void swin_block_kernel(const float* __restrict__ x,
                       const float* __restrict__ mask_matrix,   // (256,49,49)
                       const float* __restrict__ qkv_b,         // (384)
                       const float* __restrict__ proj_b,        // (128)
                       const float* __restrict__ rpb_table,     // (169,4)
                       const int*   __restrict__ rel_index,     // (49,49)
                       const _Float16* __restrict__ w16,        // qkv_w f16 (384x128), proj_w f16 at +49152
                       float* __restrict__ out) {
  extern __shared__ __align__(128) char smem[];
  _Float16* sx  = (_Float16*)(smem + SX_OFF);
  _Float16* sq  = (_Float16*)(smem + SQ_OFF);
  _Float16* sk  = (_Float16*)(smem + SK_OFF);
  _Float16* svT = (_Float16*)(smem + SVT_OFF);

  const _Float16* wq16 = w16;
  const _Float16* wp16 = w16 + 384 * 128;

  const int g  = blockIdx.x;        // window id 0..2047
  const int bb = g >> 8;            // batch
  const int wi = g & 255;           // window-in-image (mask index)
  const int wh = wi >> 4;
  const int ww = wi & 15;

  const int tid  = threadIdx.x;
  const int wave = tid >> 5;
  const int lane = tid & 31;
  const int lN   = lane & 15;
  const int lS   = lane >> 4;

  // ---------------- Phase 1: load rolled window -> LDS f16 (64x128, pad 0) ----
  for (int idx = tid; idx < N_TOK * 32; idx += 128) {
    const int t  = idx >> 5;
    const int c4 = (idx & 31) << 2;
    const int ti = t / WS, tj = t % WS;
    const int hs = (wh * WS + ti + 3) % HW;
    const int vs = (ww * WS + tj + 3) % HW;
    const float4 v = *(const float4*)(x + (((size_t)bb * HW + hs) * HW + vs) * C_DIM + c4);
    _Float16* dst = sx + t * C_DIM + c4;
    dst[0] = (_Float16)v.x; dst[1] = (_Float16)v.y;
    dst[2] = (_Float16)v.z; dst[3] = (_Float16)v.w;
  }
  for (int idx = tid; idx < 15 * C_DIM; idx += 128) sx[N_TOK * C_DIM + idx] = (_Float16)0.f;
  __syncthreads();

  // ---------------- Phase 2: QKV GEMM (64x128)x(128x384), wave = M-tile ------
  {
    v16h a0[4];
#pragma unroll
    for (int kt = 0; kt < 4; ++kt)
      a0[kt] = load_a_frag(sx + (wave * 16) * C_DIM + kt * 32, C_DIM, lane);

    for (int nt = 0; nt < 24; ++nt) {
      v8f acc = {};
#pragma unroll
      for (int kt = 0; kt < 4; ++kt) {
        const v16h bfr = load_b_frag(wq16 + (nt * 16) * C_DIM + kt * 32, C_DIM, lane);
        acc = WMMA_F16(a0[kt], bfr, acc);
      }
      const float bias = qkv_b[nt * 16 + lN];
      if (nt < 8) {            // Q (scaled)
        const int col = nt * 16 + lN;
#pragma unroll
        for (int r = 0; r < 8; ++r) {
          const int t = wave * 16 + 8 * lS + r;
          sq[t * C_DIM + col] = (_Float16)((acc[r] + bias) * QSCALE);
        }
      } else if (nt < 16) {    // K
        const int col = (nt - 8) * 16 + lN;
#pragma unroll
        for (int r = 0; r < 8; ++r) {
          const int t = wave * 16 + 8 * lS + r;
          sk[t * C_DIM + col] = (_Float16)(acc[r] + bias);
        }
      } else {                 // V, stored transposed [channel][token]
        const int ch = (nt - 16) * 16 + lN;
#pragma unroll
        for (int r = 0; r < 8; ++r) {
          const int t = wave * 16 + 8 * lS + r;
          svT[ch * 64 + t] = (_Float16)(acc[r] + bias);
        }
      }
    }
  }
  __syncthreads();

  // ---------------- Phase 3: attention, wave = head ---------------------------
  {
    const int h = wave;
    float*    S = (float*)(smem + SS_OFF) + h * 64 * 64;
    _Float16* P = (_Float16*)(smem + SP_OFF) + h * 64 * 64;

    // S = q_h @ k_h^T  (M=64, N=64, K=32)
#pragma unroll
    for (int mt = 0; mt < 4; ++mt) {
      const v16h aq = load_a_frag(sq + (mt * 16) * C_DIM + h * 32, C_DIM, lane);
#pragma unroll
      for (int nt = 0; nt < 4; ++nt) {
        const v16h bk = load_b_frag(sk + (nt * 16) * C_DIM + h * 32, C_DIM, lane);
        v8f acc = {};
        acc = WMMA_F16(aq, bk, acc);
#pragma unroll
        for (int r = 0; r < 8; ++r)
          S[(mt * 16 + 8 * lS + r) * 64 + nt * 16 + lN] = acc[r];
      }
    }
    __syncthreads();

    // softmax rows (2 rows per lane), fused rel-pos bias + shift mask
#pragma unroll
    for (int rr = 0; rr < 2; ++rr) {
      const int i = lane + 32 * rr;
      if (i < N_TOK) {
        float* Srow = S + i * 64;
        const float* mrow = mask_matrix + (size_t)wi * N_TOK * N_TOK + i * N_TOK;
        const int*   irow = rel_index + i * N_TOK;
        for (int j = 0; j < N_TOK; ++j)
          Srow[j] += rpb_table[irow[j] * NH + h] + mrow[j];
        float mx = -1e30f;
        for (int j = 0; j < N_TOK; ++j) mx = fmaxf(mx, Srow[j]);
        float sum = 0.f;
        for (int j = 0; j < N_TOK; ++j) {
          const float e = __expf(Srow[j] - mx);
          Srow[j] = e;
          sum += e;
        }
        const float inv = 1.0f / sum;
        for (int j = 0; j < N_TOK; ++j) P[i * 64 + j] = (_Float16)(Srow[j] * inv);
        for (int j = N_TOK; j < 64; ++j) P[i * 64 + j] = (_Float16)0.f;
      } else {
        for (int j = 0; j < 64; ++j) P[i * 64 + j] = (_Float16)0.f;
      }
    }
    __syncthreads();

    // O_h = P @ v_h  (M=64, N=32, K=64) -> write into sx (A operand for proj)
#pragma unroll
    for (int mt = 0; mt < 4; ++mt) {
      const v16h ap0 = load_a_frag(P + (mt * 16) * 64 + 0, 64, lane);
      const v16h ap1 = load_a_frag(P + (mt * 16) * 64 + 32, 64, lane);
#pragma unroll
      for (int nt = 0; nt < 2; ++nt) {
        const v16h bv0 = load_b_frag(svT + (h * 32 + nt * 16) * 64 + 0, 64, lane);
        const v16h bv1 = load_b_frag(svT + (h * 32 + nt * 16) * 64 + 32, 64, lane);
        v8f acc = {};
        acc = WMMA_F16(ap0, bv0, acc);
        acc = WMMA_F16(ap1, bv1, acc);
#pragma unroll
        for (int r = 0; r < 8; ++r)
          sx[(mt * 16 + 8 * lS + r) * C_DIM + h * 32 + nt * 16 + lN] = (_Float16)acc[r];
      }
    }
  }
  __syncthreads();

  // ---------------- Phase 4: proj GEMM + fused window-reverse/roll store ------
  {
    v16h ao[4];
#pragma unroll
    for (int kt = 0; kt < 4; ++kt)
      ao[kt] = load_a_frag(sx + (wave * 16) * C_DIM + kt * 32, C_DIM, lane);

    for (int nt = 0; nt < 8; ++nt) {
      v8f acc = {};
#pragma unroll
      for (int kt = 0; kt < 4; ++kt) {
        const v16h bfr = load_b_frag(wp16 + (nt * 16) * C_DIM + kt * 32, C_DIM, lane);
        acc = WMMA_F16(ao[kt], bfr, acc);
      }
      const float bias = proj_b[nt * 16 + lN];
#pragma unroll
      for (int r = 0; r < 8; ++r) {
        const int t = wave * 16 + 8 * lS + r;
        if (t < N_TOK) {
          const int ti = t / WS, tj = t % WS;
          const int hd = (wh * WS + ti + 3) % HW;   // roll(-s) then roll(+s) cancel
          const int wd = (ww * WS + tj + 3) % HW;
          out[(((size_t)bb * HW + hd) * HW + wd) * C_DIM + nt * 16 + lN] = acc[r] + bias;
        }
      }
    }
  }
}

extern "C" void kernel_launch(void* const* d_in, const int* in_sizes, int n_in,
                              void* d_out, int out_size, void* d_ws, size_t ws_size,
                              hipStream_t stream) {
  const float* x      = (const float*)d_in[0];
  const float* maskm  = (const float*)d_in[1];
  const float* qkv_w  = (const float*)d_in[2];
  const float* qkv_b  = (const float*)d_in[3];
  const float* proj_w = (const float*)d_in[4];
  const float* proj_b = (const float*)d_in[5];
  const float* rpb    = (const float*)d_in[6];
  const int*   rel    = (const int*)d_in[7];
  float* out = (float*)d_out;

  _Float16* w16 = (_Float16*)d_ws;  // 384*128 + 128*128 halves = 128 KB

  (void)in_sizes; (void)n_in; (void)out_size; (void)ws_size;

  // allow >64KB dynamic LDS
  static bool attr_set = false; // idempotent config, not state affecting work
  hipFuncSetAttribute((const void*)swin_block_kernel,
                      hipFuncAttributeMaxDynamicSharedMemorySize, SMEM_BYTES);
  (void)attr_set;

  const int wtotal = 384 * 128 + 128 * 128;
  prep_weights<<<(wtotal + 255) / 256, 256, 0, stream>>>(qkv_w, proj_w, w16);
  swin_block_kernel<<<2048, 128, SMEM_BYTES, stream>>>(
      x, maskm, qkv_b, proj_b, rpb, rel, w16, out);
}
